// HeadAttention_33655363731866
// MI455X (gfx1250) — matrix operand order repaired and checked by reading, not compiled, not run on hardware
//
#include <hip/hip_runtime.h>

// ---------------------------------------------------------------------------
// Single-head causal attention for MI455X (gfx1250), wave32 + WMMA bf16.
//   Kernel 0: weight transpose+convert f32[1024,64] -> bf16[64,1024] (x3)
//   Kernel 1: fused QKV projection -> Q (pre-scaled), K row-major bf16;
//             V transposed bf16
//   Kernel 2: flash attention; softmax denominator computed via a 5th
//             "ones-column" WMMA instead of cross-lane shuffle sums
// ---------------------------------------------------------------------------

typedef __attribute__((ext_vector_type(16))) __bf16 v16bf;
typedef __attribute__((ext_vector_type(8)))  __bf16 v8bf;
typedef __attribute__((ext_vector_type(8)))  float  v8f;

#define WMMA_BF16(a, b, c) \
  __builtin_amdgcn_wmma_f32_16x16x32_bf16(false, (a), false, (b), (short)0, (c), false, false)

constexpr int EMB   = 1024;
constexpr int HEAD  = 64;
constexpr int SEQ   = 4096;
constexpr int BATCH = 4;
constexpr int ROWS  = BATCH * SEQ;   // 16384

static __device__ __forceinline__ v16bf load16(const __bf16* p) {
  return *(const v16bf*)p;           // 32B, callers guarantee alignment
}
static __device__ __forceinline__ v8bf load8(const __bf16* p) {
  return *(const v8bf*)p;            // 16B
}
static __device__ __forceinline__ v16bf cat8(v8bf lo, v8bf hi) {
  v16bf r;
#pragma unroll
  for (int i = 0; i < 8; ++i) { r[i] = lo[i]; r[i + 8] = hi[i]; }
  return r;
}

// ---------------------------------------------------------------------------
// Kernel 0: W[1024,64] f32 -> Wt[64,1024] bf16 for Wq/Wk/Wv.
// ---------------------------------------------------------------------------
__global__ __launch_bounds__(256) void wconv_kernel(
    const float* __restrict__ Wq, const float* __restrict__ Wk,
    const float* __restrict__ Wv,
    __bf16* __restrict__ Wqt, __bf16* __restrict__ Wkt, __bf16* __restrict__ Wvt)
{
  const int idx = blockIdx.x * 256 + threadIdx.x;   // over 1024*64
  const int k = idx >> 6;
  const int n = idx & 63;
  Wqt[n * EMB + k] = (__bf16)Wq[idx];
  Wkt[n * EMB + k] = (__bf16)Wk[idx];
  Wvt[n * EMB + k] = (__bf16)Wv[idx];
}

// ---------------------------------------------------------------------------
// Kernel 1: QKV projection.
// Grid: ROWS/16 blocks, 128 threads (4 waves). Block owns 16 rows of X;
// wave w owns output columns [16w,16w+16) with three f32 accumulators.
// Q is pre-scaled by 1/sqrt(HEAD) so the attention loop never multiplies.
// ---------------------------------------------------------------------------
__global__ __launch_bounds__(128) void qkv_proj_kernel(
    const float* __restrict__ X,
    const __bf16* __restrict__ Wqt, const __bf16* __restrict__ Wkt,
    const __bf16* __restrict__ Wvt,
    const float* __restrict__ bq, const float* __restrict__ bk,
    const float* __restrict__ bv,
    __bf16* __restrict__ qws, __bf16* __restrict__ kws, __bf16* __restrict__ vt)
{
  __shared__ __attribute__((aligned(16))) float xs[16 * 32];

  const int m0   = blockIdx.x * 16;
  const int tid  = threadIdx.x;
  const int wave = tid >> 5;
  const int lane = tid & 31;
  const int l16  = lane & 15;
  const int half = lane >> 4;
  const int ncol = wave * 16 + l16;

  const __bf16* wq = Wqt + (size_t)ncol * EMB;
  const __bf16* wk = Wkt + (size_t)ncol * EMB;
  const __bf16* wv = Wvt + (size_t)ncol * EMB;

  v8f cq = {}; v8f ck = {}; v8f cv = {};

  for (int kk = 0; kk < EMB; kk += 32) {
    __syncthreads();   // WAR on xs
    {
      const int idx = tid * 4;                 // 128 thr * 4 = 512 = 16x32
      const int row = idx >> 5;
      const int col = idx & 31;
      const float4 v = *(const float4*)(X + (size_t)(m0 + row) * EMB + kk + col);
      *(float4*)(xs + row * 32 + col) = v;
    }
    __syncthreads();

    // A fragment: k = 16*half + i (contiguous)
    v16bf a;
    {
      const float* lo = xs + l16 * 32 + 16 * half;
      const float* hi = lo + 8;
#pragma unroll
      for (int i = 0; i < 8; ++i) { a[i] = (__bf16)lo[i]; a[i + 8] = (__bf16)hi[i]; }
    }

    // B fragments: contiguous 16 bf16 at k = kk + 16*half
    const v16bf bqf = load16(wq + kk + 16 * half);
    const v16bf bkf = load16(wk + kk + 16 * half);
    const v16bf bvf = load16(wv + kk + 16 * half);

    cq = WMMA_BF16(a, bqf, cq);
    ck = WMMA_BF16(a, bkf, ck);
    cv = WMMA_BF16(a, bvf, cv);
  }

  const float biasq = bq[ncol];
  const float biask = bk[ncol];
  const float biasv = bv[ncol];
  const float scale = 0.125f;   // 1/sqrt(HEAD), folded into Q
#pragma unroll
  for (int r = 0; r < 8; ++r) {
    const size_t row = (size_t)(m0 + r + 8 * half);
    qws[row * HEAD + ncol] = (__bf16)((cq[r] + biasq) * scale);
    kws[row * HEAD + ncol] = (__bf16)(ck[r] + biask);
    vt[(size_t)ncol * ROWS + row] = (__bf16)(cv[r] + biasv);  // transposed V
  }
}

// ---------------------------------------------------------------------------
// Kernel 2: causal flash attention.
// Grid: (S/64, B), 128 threads (4 waves); wave owns a 16-query tile.
// Per 32-key block: 4 score WMMAs, row-max via shfl over 16-lane halves,
// P staged through per-wave LDS (C->A relayout, in-order DS), then
// 4 P*V WMMAs + 1 "ones-column" WMMA that accumulates the softmax
// denominator in a 5th f32 fragment (no shuffle sums needed).
// ---------------------------------------------------------------------------
__global__ __launch_bounds__(128) void flash_attn_kernel(
    const __bf16* __restrict__ qws, const __bf16* __restrict__ kws,
    const __bf16* __restrict__ vt, float* __restrict__ out)
{
  __shared__ __attribute__((aligned(16))) __bf16 plds[4][16 * 32];

  const int b    = blockIdx.y;
  const int tid  = threadIdx.x;
  const int wave = tid >> 5;
  const int lane = tid & 31;
  const int l16  = lane & 15;
  const int half = lane >> 4;
  const int qt   = blockIdx.x * 4 + wave;
  const int q0   = qt * 16;

  const __bf16* Q = qws + (size_t)b * SEQ * HEAD;
  const __bf16* K = kws + (size_t)b * SEQ * HEAD;

  // Q A-fragments (hd contraction split 2x32); chunks of 8 contiguous bf16
  const __bf16* qp = Q + (size_t)(q0 + l16) * HEAD;
  const v16bf aq0 = cat8(load8(qp + 16 * half),      load8(qp + 8 + 16 * half));
  const v16bf aq1 = cat8(load8(qp + 32 + 16 * half), load8(qp + 40 + 16 * half));

  // Constant all-ones B fragment for the denominator column
  v16bf onesf;
#pragma unroll
  for (int i = 0; i < 16; ++i) onesf[i] = (__bf16)1.0f;

  float m[8];
  v8f o0 = {}, o1 = {}, o2 = {}, o3 = {}, o4 = {};   // o4 = running row sums
#pragma unroll
  for (int r = 0; r < 8; ++r) m[r] = -1e30f;

  const int nkb = qt / 2 + 1;

  for (int kb = 0; kb < nkb; ++kb) {
    const int key0 = kb * 32;

    // ---- scores: two 16x16 tiles (keys key0..+15 and key0+16..+31)
    const __bf16* kp0 = K + (size_t)(key0 + l16) * HEAD;
    const __bf16* kp1 = K + (size_t)(key0 + 16 + l16) * HEAD;
    v8f s0 = {}, s1 = {};
    s0 = WMMA_BF16(aq0, load16(kp0 + 16 * half), s0);
    s1 = WMMA_BF16(aq0, load16(kp1 + 16 * half), s1);
    s0 = WMMA_BF16(aq1, load16(kp0 + 32 + 16 * half), s0);
    s1 = WMMA_BF16(aq1, load16(kp1 + 32 + 16 * half), s1);

    // ---- mask + online max (row = r+8*half striped over 16 lanes)
    float corr[8];
#pragma unroll
    for (int r = 0; r < 8; ++r) {
      const int q = q0 + r + 8 * half;
      const float v0 = (key0 + l16      <= q) ? s0[r] : -1e30f;
      const float v1 = (key0 + 16 + l16 <= q) ? s1[r] : -1e30f;

      float mx = fmaxf(v0, v1);
#pragma unroll
      for (int d = 1; d < 16; d <<= 1) mx = fmaxf(mx, __shfl_xor(mx, d, 32));

      const float mn = fmaxf(m[r], mx);
      corr[r] = __expf(m[r] - mn);
      m[r] = mn;

      plds[wave][(r + 8 * half) * 32 + l16]      = (__bf16)__expf(v0 - mn);
      plds[wave][(r + 8 * half) * 32 + 16 + l16] = (__bf16)__expf(v1 - mn);
    }

    // ---- rescale running output (and denominator) by the max correction
#pragma unroll
    for (int r = 0; r < 8; ++r) {
      o0[r] *= corr[r]; o1[r] *= corr[r]; o2[r] *= corr[r];
      o3[r] *= corr[r]; o4[r] *= corr[r];
    }

    // ---- P as A-fragment: two 16B LDS reads (in-order DS within wave)
    const __bf16* pr = &plds[wave][l16 * 32];
    const v16bf ap = cat8(load8(pr + 16 * half), load8(pr + 8 + 16 * half));

    // ---- V B-fragments from transposed V: contiguous over key index
    const __bf16* vb = vt + (size_t)b * SEQ + key0 + 16 * half;
    const v16bf bv0 = load16(vb + (size_t)(l16)      * ROWS);
    const v16bf bv1 = load16(vb + (size_t)(16 + l16) * ROWS);
    const v16bf bv2 = load16(vb + (size_t)(32 + l16) * ROWS);
    const v16bf bv3 = load16(vb + (size_t)(48 + l16) * ROWS);

    o0 = WMMA_BF16(ap, bv0, o0);
    o1 = WMMA_BF16(ap, bv1, o1);
    o2 = WMMA_BF16(ap, bv2, o2);
    o3 = WMMA_BF16(ap, bv3, o3);
    o4 = WMMA_BF16(ap, onesf, o4);   // row sums of P -> denominator
  }

  // ---- normalize and store f32 output [B, S, H]
#pragma unroll
  for (int r = 0; r < 8; ++r) {
    const float inv = 1.0f / o4[r];
    float* op = out + ((size_t)b * SEQ + q0 + r + 8 * half) * HEAD;
    op[0  + l16] = o0[r] * inv;
    op[16 + l16] = o1[r] * inv;
    op[32 + l16] = o2[r] * inv;
    op[48 + l16] = o3[r] * inv;
  }
}

// ---------------------------------------------------------------------------
extern "C" void kernel_launch(void* const* d_in, const int* in_sizes, int n_in,
                              void* d_out, int out_size, void* d_ws, size_t ws_size,
                              hipStream_t stream) {
  const float* X  = (const float*)d_in[0];
  const float* Wk = (const float*)d_in[1];
  const float* bk = (const float*)d_in[2];
  const float* Wq = (const float*)d_in[3];
  const float* bq = (const float*)d_in[4];
  const float* Wv = (const float*)d_in[5];
  const float* bv = (const float*)d_in[6];
  float* out = (float*)d_out;

  const size_t qkv_elems = (size_t)ROWS * HEAD;          // 1 Mi elems
  __bf16* qws = (__bf16*)d_ws;                           // 2 MB
  __bf16* kws = qws + qkv_elems;                         // 2 MB
  __bf16* vt  = kws + qkv_elems;                         // 2 MB (transposed V)
  __bf16* Wqt = vt + qkv_elems;                          // 128 KB
  __bf16* Wkt = Wqt + (size_t)EMB * HEAD;
  __bf16* Wvt = Wkt + (size_t)EMB * HEAD;

  wconv_kernel<<<dim3((EMB * HEAD) / 256), dim3(256), 0, stream>>>(
      Wq, Wk, Wv, Wqt, Wkt, Wvt);

  qkv_proj_kernel<<<dim3(ROWS / 16), dim3(128), 0, stream>>>(
      X, Wqt, Wkt, Wvt, bq, bk, bv, qws, kws, vt);

  flash_attn_kernel<<<dim3(SEQ / 64, BATCH), dim3(128), 0, stream>>>(
      qws, kws, vt, out);
}
